// RNN_63995012710975
// MI455X (gfx1250) — matrix operand order repaired and checked by reading, not compiled, not run on hardware
//
#include <hip/hip_runtime.h>

typedef unsigned short u16;
typedef unsigned int   u32;
typedef __attribute__((ext_vector_type(16))) __bf16 v16bf;
typedef __attribute__((ext_vector_type(8)))  float  v8f;

union Frag { v16bf v; uint4 q[2]; };

__device__ __forceinline__ u16 f2bf(float f) {
    union { float f; u32 u; } c; c.f = f;
    u32 r = c.u + 0x7FFFu + ((c.u >> 16) & 1u);   // round-to-nearest-even
    return (u16)(r >> 16);
}
__device__ __forceinline__ float bf2f(u16 h) {
    union { u32 u; float f; } c; c.u = ((u32)h) << 16;
    return c.f;
}
__device__ __forceinline__ v8f wmma_bf16(v16bf a, v16bf b, v8f c) {
    return __builtin_amdgcn_wmma_f32_16x16x32_bf16(
        false, a, false, b, (short)0, c, false, false);
}

// 32-byte global -> LDS async copy (2 x b128 sharing one address pair;
// INST_OFFSET applies to both LDS and global addresses per ISA 10.x).
__device__ __forceinline__ void async_ld_32B(u32 lds_off, const void* g) {
    asm volatile("global_load_async_to_lds_b128 %0, %1, off\n\t"
                 "global_load_async_to_lds_b128 %0, %1, off offset:16"
                 :: "v"(lds_off), "v"(g) : "memory");
}
__device__ __forceinline__ void wait_async0() {
    asm volatile("s_wait_asynccnt 0" ::: "memory");
}
__device__ __forceinline__ u32 lds_off32(const void* p) {
    return (u32)(size_t)p;   // generic LDS addr: low 32 bits = LDS byte offset
}

// ---------------------------------------------------------------- conversions
__global__ void __launch_bounds__(256) cvt_f32_bf16_k(
    const float* __restrict__ src, u16* __restrict__ dst, int n)
{
    int i = (blockIdx.x * 256 + threadIdx.x) * 8;
    if (i + 8 > n) return;
    float4 a = *(const float4*)(src + i);
    float4 b = *(const float4*)(src + i + 4);
    uint4 o;
    o.x = (u32)f2bf(a.x) | ((u32)f2bf(a.y) << 16);
    o.y = (u32)f2bf(a.z) | ((u32)f2bf(a.w) << 16);
    o.z = (u32)f2bf(b.x) | ((u32)f2bf(b.y) << 16);
    o.w = (u32)f2bf(b.z) | ((u32)f2bf(b.w) << 16);
    *(uint4*)(dst + i) = o;
}

__global__ void __launch_bounds__(256) zero16_k(uint4* __restrict__ p, int n16)
{
    int i = blockIdx.x * 256 + threadIdx.x;
    if (i < n16) p[i] = make_uint4(0u, 0u, 0u, 0u);
}

// ------------------------------------------------------ generic bf16 GEMM
// out[M,N](bf16) = act(A[M,K](bf16) @ W[N,K]^T(bf16) + bias[N](f32))
// Block tile 128x128, BK=32, 256 threads (8 waves: 2 along M x 4 along N).
// Double-buffered LDS fed by async global->LDS copies (ASYNCcnt).
__global__ void __launch_bounds__(256) gemm_bf16_k(
    const u16* __restrict__ A, const u16* __restrict__ W,
    const float* __restrict__ bias, u16* __restrict__ out,
    int M, int N, int K, int relu)
{
    __shared__ u16 sA[2][128][40];
    __shared__ u16 sB[2][128][40];

    const int tid  = threadIdx.x;
    const int lane = tid & 31;
    const int wave = tid >> 5;
    const int wm   = wave & 1;      // 64 rows each
    const int wn   = wave >> 1;     // 32 cols each
    const int lh   = lane >> 4;     // lane half
    const int l16  = lane & 15;
    const int m0   = blockIdx.x * 128;
    const int n0   = blockIdx.y * 128;

    const int srow = tid >> 1;
    const int scol = (tid & 1) * 16;
    const u16* Ag = A + (size_t)(m0 + srow) * K + scol;
    const u16* Wg = W + (size_t)(n0 + srow) * K + scol;
    const u32 ldsA[2] = { lds_off32(&sA[0][srow][scol]), lds_off32(&sA[1][srow][scol]) };
    const u32 ldsB[2] = { lds_off32(&sB[0][srow][scol]), lds_off32(&sB[1][srow][scol]) };

    v8f acc[4][2];
    v8f zf = {};
    for (int i = 0; i < 4; ++i)
        for (int j = 0; j < 2; ++j) acc[i][j] = zf;

    // prologue: stage tile 0 into buffer 0
    async_ld_32B(ldsA[0], Ag);
    async_ld_32B(ldsB[0], Wg);
    wait_async0();
    __syncthreads();

    const int nk = K >> 5;
    for (int i = 0; i < nk; ++i) {
        const int buf = i & 1;
        if (i + 1 < nk) {                    // prefetch next tile
            async_ld_32B(ldsA[buf ^ 1], Ag + (i + 1) * 32);
            async_ld_32B(ldsB[buf ^ 1], Wg + (i + 1) * 32);
        }

        Frag af[4], bfr[2];
        #pragma unroll
        for (int mt = 0; mt < 4; ++mt) {
            const int r = wm * 64 + mt * 16 + l16;
            af[mt].q[0] = *(const uint4*)&sA[buf][r][8 * lh];       // K = 8h + e
            af[mt].q[1] = *(const uint4*)&sA[buf][r][16 + 8 * lh];  // K = 16+8h+e
        }
        #pragma unroll
        for (int nt = 0; nt < 2; ++nt) {
            const int r = wn * 32 + nt * 16 + l16;
            bfr[nt].q[0] = *(const uint4*)&sB[buf][r][8 * lh];      // K = 8h + e
            bfr[nt].q[1] = *(const uint4*)&sB[buf][r][16 + 8 * lh];
        }
        #pragma unroll
        for (int mt = 0; mt < 4; ++mt)
            #pragma unroll
            for (int nt = 0; nt < 2; ++nt)
                acc[mt][nt] = wmma_bf16(af[mt].v, bfr[nt].v, acc[mt][nt]);

        wait_async0();      // next tile landed (own wave)
        __syncthreads();    // all waves: next buf ready, current buf consumed
    }

    #pragma unroll
    for (int nt = 0; nt < 2; ++nt) {
        const int col = n0 + wn * 32 + nt * 16 + l16;
        const float bv = bias[col];
        #pragma unroll
        for (int mt = 0; mt < 4; ++mt) {
            const int rowb = m0 + wm * 64 + mt * 16 + 8 * lh;
            #pragma unroll
            for (int r = 0; r < 8; ++r) {
                float v = acc[mt][nt][r] + bv;
                if (relu) v = fmaxf(v, 0.0f);
                out[(size_t)(rowb + r) * N + col] = f2bf(v);
            }
        }
    }
}

// ------------------------------------------------------------- GRU step
// For one timestep: gh = h_prev @ W_hh^T (+b_hh), fuse gates with gx[t],
// write h_new (f32 ping-pong) and hs[t] (bf16, next step's A operand).
// Block: 128 batch rows x 64 hidden cols x 3 gates. 8 waves: 4(M) x 2(J).
__global__ void __launch_bounds__(256) gru_step_k(
    const u16*  __restrict__ gx,       // (2048,3072) bf16 (b_ih included)
    const u16*  __restrict__ hprev_b,  // (2048,1024) bf16
    const float* __restrict__ hprev_f, // (2048,1024) f32
    float*      __restrict__ hcur_f,   // (2048,1024) f32
    u16*        __restrict__ hs_t,     // (2048,1024) bf16
    const u16*  __restrict__ Whh,      // (3072,1024) bf16
    const float* __restrict__ bhh)     // (3072) f32
{
    __shared__ u16 sH[2][128][40];
    __shared__ u16 sW[2][192][40];

    const int tid  = threadIdx.x;
    const int lane = tid & 31;
    const int wave = tid >> 5;
    const int wmr  = wave & 3;      // 32 rows each
    const int wjc  = wave >> 2;     // 32 cols each
    const int lh   = lane >> 4;
    const int l16  = lane & 15;
    const int m0   = blockIdx.x * 128;
    const int j0   = blockIdx.y * 64;

    const int srow = tid >> 1;
    const int scol = (tid & 1) * 16;
    const u16* Hg = hprev_b + (size_t)(m0 + srow) * 1024 + scol;

    const int rr0 = tid >> 1;                 // W staging task 0 (rows 0..127)
    const bool t2 = tid < 128;                // task 1 covers rows 128..191 (wave-uniform)
    const int rr1 = 128 + (tid >> 1);
    const u16* Wg0 = Whh + (size_t)((rr0 >> 6) * 1024 + j0 + (rr0 & 63)) * 1024 + scol;
    const u16* Wg1 = Whh + (size_t)((rr1 >> 6) * 1024 + j0 + (rr1 & 63)) * 1024 + scol;

    const u32 ldsH[2]  = { lds_off32(&sH[0][srow][scol]), lds_off32(&sH[1][srow][scol]) };
    const u32 ldsW0[2] = { lds_off32(&sW[0][rr0][scol]),  lds_off32(&sW[1][rr0][scol]) };
    const u32 ldsW1[2] = { lds_off32(&sW[0][rr1][scol]),  lds_off32(&sW[1][rr1][scol]) };

    v8f acc[3][2][2];
    v8f zf = {};
    for (int g = 0; g < 3; ++g)
        for (int i = 0; i < 2; ++i)
            for (int j = 0; j < 2; ++j) acc[g][i][j] = zf;

    // prologue: stage tile 0
    async_ld_32B(ldsH[0], Hg);
    async_ld_32B(ldsW0[0], Wg0);
    if (t2) async_ld_32B(ldsW1[0], Wg1);
    wait_async0();
    __syncthreads();

    for (int i = 0; i < 32; ++i) {           // K = 1024 = 32 tiles
        const int buf = i & 1;
        if (i + 1 < 32) {
            const int kn = (i + 1) * 32;
            async_ld_32B(ldsH[buf ^ 1], Hg + kn);
            async_ld_32B(ldsW0[buf ^ 1], Wg0 + kn);
            if (t2) async_ld_32B(ldsW1[buf ^ 1], Wg1 + kn);
        }

        Frag af[2];
        #pragma unroll
        for (int mt = 0; mt < 2; ++mt) {
            const int r = wmr * 32 + mt * 16 + l16;
            af[mt].q[0] = *(const uint4*)&sH[buf][r][8 * lh];
            af[mt].q[1] = *(const uint4*)&sH[buf][r][16 + 8 * lh];
        }
        #pragma unroll
        for (int g = 0; g < 3; ++g) {
            Frag bfr[2];
            #pragma unroll
            for (int nt = 0; nt < 2; ++nt) {
                const int r = g * 64 + wjc * 32 + nt * 16 + l16;
                bfr[nt].q[0] = *(const uint4*)&sW[buf][r][8 * lh];
                bfr[nt].q[1] = *(const uint4*)&sW[buf][r][16 + 8 * lh];
            }
            #pragma unroll
            for (int mt = 0; mt < 2; ++mt)
                #pragma unroll
                for (int nt = 0; nt < 2; ++nt)
                    acc[g][mt][nt] = wmma_bf16(af[mt].v, bfr[nt].v, acc[g][mt][nt]);
        }

        wait_async0();
        __syncthreads();
    }

    #pragma unroll
    for (int nt = 0; nt < 2; ++nt) {
        const int j = j0 + wjc * 32 + nt * 16 + l16;
        const float br = bhh[j];
        const float bz = bhh[1024 + j];
        const float bn = bhh[2048 + j];
        #pragma unroll
        for (int mt = 0; mt < 2; ++mt) {
            const int rowb = m0 + wmr * 32 + mt * 16 + 8 * lh;
            #pragma unroll
            for (int r = 0; r < 8; ++r) {
                const int b = rowb + r;
                const size_t gb = (size_t)b * 3072 + j;
                const float xr = bf2f(gx[gb]);
                const float xz = bf2f(gx[gb + 1024]);
                const float xn = bf2f(gx[gb + 2048]);
                const float hr = acc[0][mt][nt][r] + br;
                const float hz = acc[1][mt][nt][r] + bz;
                const float hn = acc[2][mt][nt][r] + bn;
                const float rg = 1.0f / (1.0f + __expf(-(xr + hr)));
                const float zg = 1.0f / (1.0f + __expf(-(xz + hz)));
                const float ng = tanhf(xn + rg * hn);
                const float hp = hprev_f[(size_t)b * 1024 + j];
                const float hv = (1.0f - zg) * ng + zg * hp;
                hcur_f[(size_t)b * 1024 + j] = hv;
                hs_t[(size_t)b * 1024 + j] = f2bf(hv);
            }
        }
    }
}

// --------------------------------------------------------- output projection
// out[i] = dot(hs[i,:], W3) + b3 ; flat i == t*B+b matches the raw reshape.
__global__ void __launch_bounds__(256) proj_k(
    const u16* __restrict__ hs, const float* __restrict__ W3,
    const float* __restrict__ b3, float* __restrict__ out)
{
    const int lane = threadIdx.x & 31;
    const int row  = blockIdx.x * 8 + (threadIdx.x >> 5);
    const u16* hp = hs + (size_t)row * 1024;
    float s = 0.0f;
    for (int k = lane * 4; k < 1024; k += 128) {
        uint2  h = *(const uint2*)(hp + k);
        float4 w = *(const float4*)(W3 + k);
        s += bf2f((u16)(h.x & 0xFFFFu)) * w.x;
        s += bf2f((u16)(h.x >> 16))     * w.y;
        s += bf2f((u16)(h.y & 0xFFFFu)) * w.z;
        s += bf2f((u16)(h.y >> 16))     * w.w;
    }
    #pragma unroll
    for (int off = 16; off > 0; off >>= 1) s += __shfl_xor(s, off, 32);
    if (lane == 0) out[row] = s + b3[0];
}

// ---------------------------------------------------------------------------
extern "C" void kernel_launch(void* const* d_in, const int* in_sizes, int n_in,
                              void* d_out, int out_size, void* d_ws, size_t ws_size,
                              hipStream_t stream)
{
    (void)in_sizes; (void)n_in; (void)out_size; (void)ws_size;
    const float* inputs = (const float*)d_in[0];
    const float* W1     = (const float*)d_in[1];
    const float* b1     = (const float*)d_in[2];
    const float* W_ih   = (const float*)d_in[3];
    const float* b_ih   = (const float*)d_in[4];
    const float* W_hh   = (const float*)d_in[5];
    const float* b_hh   = (const float*)d_in[6];
    const float* W3     = (const float*)d_in[7];
    const float* b3     = (const float*)d_in[8];
    float* out = (float*)d_out;

    constexpr int    T = 64, B = 2048, IN = 512, H = 1024, G3H = 3 * H;
    constexpr size_t M = (size_t)T * B;   // 131072 rows

    char* w = (char*)d_ws;
    auto carve = [&](size_t bytes) {
        char* p = w;
        w += (bytes + 255) & ~(size_t)255;
        return p;
    };
    u16*   inb  = (u16*)carve(M * IN * 2);            // inputs bf16
    u16*   W1b  = (u16*)carve((size_t)H * IN * 2);
    u16*   Wihb = (u16*)carve((size_t)G3H * H * 2);
    u16*   Whhb = (u16*)carve((size_t)G3H * H * 2);
    u16*   xb   = (u16*)carve(M * H * 2);             // relu(fc1) bf16
    u16*   gxb  = (u16*)carve(M * G3H * 2);           // gx_all bf16
    u16*   hs   = (u16*)carve(M * H * 2);             // hidden states bf16
    u16*   h0b  = (u16*)carve((size_t)B * H * 2);     // zero h0 bf16
    float* hA   = (float*)carve((size_t)B * H * 4);   // h ping
    float* hB   = (float*)carve((size_t)B * H * 4);   // h pong

    // fp32 -> bf16 staging
    cvt_f32_bf16_k<<<(int)(M * IN / 2048), 256, 0, stream>>>(inputs, inb, (int)(M * IN));
    cvt_f32_bf16_k<<<(H * IN) / 2048, 256, 0, stream>>>(W1, W1b, H * IN);
    cvt_f32_bf16_k<<<(G3H * H) / 2048, 256, 0, stream>>>(W_ih, Wihb, G3H * H);
    cvt_f32_bf16_k<<<(G3H * H) / 2048, 256, 0, stream>>>(W_hh, Whhb, G3H * H);
    zero16_k<<<(B * H * 2 / 16 + 255) / 256, 256, 0, stream>>>((uint4*)h0b, B * H * 2 / 16);
    zero16_k<<<(B * H * 4 / 16 + 255) / 256, 256, 0, stream>>>((uint4*)hA, B * H * 4 / 16);

    // x = relu(inputs @ W1^T + b1)
    gemm_bf16_k<<<dim3((unsigned)(M / 128), H / 128), 256, 0, stream>>>(
        inb, W1b, b1, xb, (int)M, H, IN, 1);
    // gx_all = x @ W_ih^T + b_ih
    gemm_bf16_k<<<dim3((unsigned)(M / 128), G3H / 128), 256, 0, stream>>>(
        xb, Wihb, b_ih, gxb, (int)M, G3H, H, 0);

    // sequential GRU scan
    for (int t = 0; t < T; ++t) {
        const u16*   hprevb = (t == 0) ? h0b : hs + (size_t)(t - 1) * B * H;
        const float* hpf    = (t & 1) ? hB : hA;
        float*       hcf    = (t & 1) ? hA : hB;
        gru_step_k<<<dim3(B / 128, H / 64), 256, 0, stream>>>(
            gxb + (size_t)t * B * G3H, hprevb, hpf, hcf,
            hs + (size_t)t * B * H, Whhb, b_hh);
    }

    // q = hs @ W3^T + b3  (flat layout == reference's raw reshape)
    proj_k<<<(unsigned)(M / 8), 256, 0, stream>>>(hs, W3, b3, out);
}
